// WhisperMQAttention_19481971654968
// MI455X (gfx1250) — matrix operand order repaired and checked by reading, not compiled, not run on hardware
//
#include <hip/hip_runtime.h>
#include <hip/hip_bf16.h>

#define EMBED   1024
#define NHEADS  16
#define HDIM    64
#define BSZ     2
#define SEQ     2048
#define MTOT    (BSZ * SEQ)      // 4096 rows of hidden
#define QSCALE  0.125f           // 64^-0.5

typedef __attribute__((ext_vector_type(16))) __bf16 v16bf;
typedef __attribute__((ext_vector_type(8)))  __bf16 v8bf;
typedef __attribute__((ext_vector_type(8)))  float  v8f;
typedef __attribute__((ext_vector_type(4)))  float  v4f;

// ---------------------------------------------------------------------------
// Fragment loaders for V_WMMA_F32_16X16X32_BF16 (wave32).
// A matrix 16x32 (MxK), 16-bit: lane L holds row M=L&15.
//   lanes 0-15:  VGPR0-3 = K 0..7,  VGPR4-7 = K 16..23
//   lanes 16-31: VGPR0-3 = K 8..15, VGPR4-7 = K 24..31
static __device__ __forceinline__ v16bf load_frag_a(const __bf16* p) {
  const int koff = (threadIdx.x & 16) ? 8 : 0;
  union { v16bf v; v8bf h[2]; } u;
  u.h[0] = *(const v8bf*)(p + koff);
  u.h[1] = *(const v8bf*)(p + 16 + koff);
  return u.v;
}

// B matrix 32x16 (KxN), 16-bit: lane L holds column N=L&15.
//   lanes 0-15: K = 0..7, 16..23; lanes 16-31: K = 8..15, 24..31.
static __device__ __forceinline__ v16bf load_frag_b(const __bf16* p) {
  const int kb = (threadIdx.x & 16) ? 8 : 0;
  union { v16bf v; v8bf h[2]; } u;
  u.h[0] = *(const v8bf*)(p + kb);
  u.h[1] = *(const v8bf*)(p + 16 + kb);
  return u.v;
}

// ---------------------------------------------------------------------------
// fp32 -> bf16, 8 elements per thread (b128 in, b128 out). n % 8 == 0.
__global__ void f32_to_bf16_kernel(const float* __restrict__ in,
                                   __bf16* __restrict__ out, int n) {
  int i = (blockIdx.x * blockDim.x + threadIdx.x) * 8;
  if (i < n) {
    v4f a = *(const v4f*)(in + i);
    v4f b = *(const v4f*)(in + i + 4);
    v8bf r;
#pragma unroll
    for (int j = 0; j < 4; ++j) { r[j] = (__bf16)a[j]; r[4 + j] = (__bf16)b[j]; }
    *(v8bf*)(out + i) = r;
  }
}

// v [MTOT, HDIM] -> vT [BSZ, HDIM, SEQ]
__global__ void transpose_v_kernel(const __bf16* __restrict__ v,
                                   __bf16* __restrict__ vT) {
  int i = blockIdx.x * blockDim.x + threadIdx.x;
  if (i < MTOT * HDIM) {
    int tg = i / HDIM, d = i % HDIM;
    int b = tg / SEQ, t = tg % SEQ;
    vT[((size_t)b * HDIM + d) * SEQ + t] = v[i];
  }
}

// ---------------------------------------------------------------------------
// Generic GEMM: out[M,N] = (A[M,K] @ W[N,K]^T + bias) * scale
// A, W bf16 row-major. One wave computes a 64x64 tile:
// 16 WMMAs per K-step (K-step 32), software double-buffered fragments.
__global__ __launch_bounds__(32)
void gemm_wmma_bf16(const __bf16* __restrict__ A, const __bf16* __restrict__ W,
                    const float* __restrict__ bias, float scale,
                    __bf16* __restrict__ outb, float* __restrict__ outf,
                    int M, int N, int K) {
  const int lane = threadIdx.x & 31;
  const int hi   = lane >> 4;
  const int nl   = lane & 15;
  const int m0   = blockIdx.y << 6;
  const int n0   = blockIdx.x << 6;

  const __bf16* arow[4];
  const __bf16* brow[4];
#pragma unroll
  for (int i = 0; i < 4; ++i) arow[i] = A + (size_t)(m0 + i * 16 + nl) * K;
#pragma unroll
  for (int j = 0; j < 4; ++j) brow[j] = W + (size_t)(n0 + j * 16 + nl) * K;

  v8f acc[16] = {};
  v16bf af[4], bm[4];
#pragma unroll
  for (int i = 0; i < 4; ++i) af[i] = load_frag_a(arow[i]);
#pragma unroll
  for (int j = 0; j < 4; ++j) bm[j] = load_frag_b(brow[j]);

  int k0 = 0;
  for (; k0 < K - 32; k0 += 32) {
    // prefetch-ahead hint for the streamed rows (global_prefetch_b8)
    __builtin_prefetch(arow[0] + k0 + 512, 0, 1);
    __builtin_prefetch(brow[0] + k0 + 512, 0, 1);
    // next K-step fragments issued before this step's WMMAs
    v16bf afn[4], bmn[4];
#pragma unroll
    for (int i = 0; i < 4; ++i) afn[i] = load_frag_a(arow[i] + k0 + 32);
#pragma unroll
    for (int j = 0; j < 4; ++j) bmn[j] = load_frag_b(brow[j] + k0 + 32);
#pragma unroll
    for (int i = 0; i < 4; ++i)
#pragma unroll
      for (int j = 0; j < 4; ++j)
        acc[i * 4 + j] = __builtin_amdgcn_wmma_f32_16x16x32_bf16(
            false, af[i], false, bm[j], (short)0, acc[i * 4 + j], false, false);
#pragma unroll
    for (int i = 0; i < 4; ++i) af[i] = afn[i];
#pragma unroll
    for (int j = 0; j < 4; ++j) bm[j] = bmn[j];
  }
#pragma unroll
  for (int i = 0; i < 4; ++i)
#pragma unroll
    for (int j = 0; j < 4; ++j)
      acc[i * 4 + j] = __builtin_amdgcn_wmma_f32_16x16x32_bf16(
          false, af[i], false, bm[j], (short)0, acc[i * 4 + j], false, false);

#pragma unroll
  for (int j = 0; j < 4; ++j) {
    int n = n0 + j * 16 + nl;
    float bv = bias ? bias[n] : 0.0f;
#pragma unroll
    for (int i = 0; i < 4; ++i) {
#pragma unroll
      for (int r = 0; r < 8; ++r) {
        int m = m0 + i * 16 + r + (hi << 3);
        float val = (acc[i * 4 + j][r] + bv) * scale;
        if (outf) outf[(size_t)m * N + n] = val;
        else      outb[(size_t)m * N + n] = (__bf16)val;
      }
    }
  }
}

// ---------------------------------------------------------------------------
// Flash-style MQA attention. One wave per (b, h, 32-query-row block).
// Q already scaled. K bf16 [B*S, 64] row-major; VT bf16 [B, 64, S].
// K/V tile fragments are shared across the two 16-row query blocks:
// 16 WMMAs per 32-key tile.
__global__ __launch_bounds__(32)
void mqa_attention_kernel(const __bf16* __restrict__ Q,
                          const __bf16* __restrict__ Kc,
                          const __bf16* __restrict__ VT,
                          __bf16* __restrict__ CTX) {
  __shared__ __bf16 plds[32 * 40];  // 32 rows x 32 probs, pitch 40 (80B)
  const int lane = threadIdx.x & 31;
  const int hi   = lane >> 4;
  const int nl   = lane & 15;
  const int t0   = blockIdx.x << 5;
  const int h    = blockIdx.y;
  const int b    = blockIdx.z;

  v16bf qf[2][2];
#pragma unroll
  for (int qb = 0; qb < 2; ++qb) {
    const __bf16* qrow =
        Q + (size_t)(b * SEQ + t0 + qb * 16 + nl) * EMBED + h * HDIM;
    qf[qb][0] = load_frag_a(qrow);
    qf[qb][1] = load_frag_a(qrow + 32);
  }

  const __bf16* kbase = Kc + (size_t)b * SEQ * HDIM;
  const __bf16* vbase = VT + (size_t)b * HDIM * SEQ;

  v8f o[2][4] = {};
  float rmax[2][8], rsum[2][8];
#pragma unroll
  for (int qb = 0; qb < 2; ++qb)
#pragma unroll
    for (int r = 0; r < 8; ++r) { rmax[qb][r] = -1e30f; rsum[qb][r] = 0.0f; }

  for (int s0 = 0; s0 < SEQ; s0 += 32) {
    __builtin_prefetch(kbase + (size_t)(s0 + 64 + nl) * HDIM, 0, 1);
    // K-tile fragments (shared by both query blocks)
    v16bf kb0[2], kb1[2];
#pragma unroll
    for (int nt = 0; nt < 2; ++nt) {
      const __bf16* krow = kbase + (size_t)(s0 + nt * 16 + nl) * HDIM;
      kb0[nt] = load_frag_b(krow);
      kb1[nt] = load_frag_b(krow + 32);
    }
    // scores: 2 query blocks x 2 key tiles, K-dim 64 = 2 WMMAs each
    v8f sc[2][2];
#pragma unroll
    for (int qb = 0; qb < 2; ++qb)
#pragma unroll
      for (int nt = 0; nt < 2; ++nt) {
        v8f z = {};
        z = __builtin_amdgcn_wmma_f32_16x16x32_bf16(
            false, qf[qb][0], false, kb0[nt], (short)0, z, false, false);
        z = __builtin_amdgcn_wmma_f32_16x16x32_bf16(
            false, qf[qb][1], false, kb1[nt], (short)0, z, false, false);
        sc[qb][nt] = z;
      }

    // Online softmax per query block; row M = r + 8*hi across a lane half.
#pragma unroll
    for (int qb = 0; qb < 2; ++qb) {
      float corr[8];
#pragma unroll
      for (int r = 0; r < 8; ++r) {
        float mx = fmaxf(sc[qb][0][r], sc[qb][1][r]);
        mx = fmaxf(mx, __shfl_xor(mx, 1, 16));
        mx = fmaxf(mx, __shfl_xor(mx, 2, 16));
        mx = fmaxf(mx, __shfl_xor(mx, 4, 16));
        mx = fmaxf(mx, __shfl_xor(mx, 8, 16));
        float nm = fmaxf(rmax[qb][r], mx);
        corr[r] = __expf(rmax[qb][r] - nm);
        rmax[qb][r] = nm;
      }
#pragma unroll
      for (int r = 0; r < 8; ++r) {
        float p0 = __expf(sc[qb][0][r] - rmax[qb][r]);
        float p1 = __expf(sc[qb][1][r] - rmax[qb][r]);
        int row = qb * 16 + r + (hi << 3);
        plds[row * 40 + nl]      = (__bf16)p0;
        plds[row * 40 + 16 + nl] = (__bf16)p1;
        float ls = p0 + p1;
        ls += __shfl_xor(ls, 1, 16);
        ls += __shfl_xor(ls, 2, 16);
        ls += __shfl_xor(ls, 4, 16);
        ls += __shfl_xor(ls, 8, 16);
        rsum[qb][r] = rsum[qb][r] * corr[r] + ls;
#pragma unroll
        for (int t = 0; t < 4; ++t) o[qb][t][r] *= corr[r];
      }
    }
    __syncthreads();  // LDS writes visible for lane-transposed read

    // Re-read probs as A-fragments (16x32) from LDS
    v16bf pf[2];
#pragma unroll
    for (int qb = 0; qb < 2; ++qb)
      pf[qb] = load_frag_a(&plds[(qb * 16 + nl) * 40]);

    // P @ V: V fragments shared by both query blocks
#pragma unroll
    for (int dt = 0; dt < 4; ++dt) {
      const __bf16* vrow = vbase + (size_t)(dt * 16 + nl) * SEQ + s0;
      v16bf vf = load_frag_b(vrow);
#pragma unroll
      for (int qb = 0; qb < 2; ++qb)
        o[qb][dt] = __builtin_amdgcn_wmma_f32_16x16x32_bf16(
            false, pf[qb], false, vf, (short)0, o[qb][dt], false, false);
    }
    __syncthreads();  // done with LDS tile before next iteration rewrites it
  }

#pragma unroll
  for (int qb = 0; qb < 2; ++qb)
#pragma unroll
    for (int dt = 0; dt < 4; ++dt) {
      int d = h * HDIM + dt * 16 + nl;
#pragma unroll
      for (int r = 0; r < 8; ++r) {
        int t = t0 + qb * 16 + r + (hi << 3);
        CTX[(size_t)(b * SEQ + t) * EMBED + d] =
            (__bf16)(o[qb][dt][r] / rsum[qb][r]);
      }
    }
}

// ---------------------------------------------------------------------------
extern "C" void kernel_launch(void* const* d_in, const int* in_sizes, int n_in,
                              void* d_out, int out_size, void* d_ws, size_t ws_size,
                              hipStream_t stream) {
  (void)in_sizes; (void)n_in; (void)out_size; (void)ws_size;

  const float* x  = (const float*)d_in[0];  // [B,T,E]
  const float* Wq = (const float*)d_in[1];  // [E,E]
  const float* bq = (const float*)d_in[2];  // [E]
  const float* Wk = (const float*)d_in[3];  // [d,E]
  const float* Wv = (const float*)d_in[4];  // [d,E]
  const float* bv = (const float*)d_in[5];  // [d]
  const float* Wo = (const float*)d_in[6];  // [E,E]
  const float* bo = (const float*)d_in[7];  // [E]
  float* out = (float*)d_out;               // [B,T,E] f32

  char* ws = (char*)d_ws;
  size_t off = 0;
  auto alloc = [&](size_t bytes) -> char* {
    char* p = ws + off;
    off += (bytes + 255) & ~((size_t)255);
    return p;
  };

  __bf16* xb   = (__bf16*)alloc((size_t)MTOT * EMBED * 2);
  __bf16* wqb  = (__bf16*)alloc((size_t)EMBED * EMBED * 2);
  __bf16* wkb  = (__bf16*)alloc((size_t)HDIM * EMBED * 2);
  __bf16* wvb  = (__bf16*)alloc((size_t)HDIM * EMBED * 2);
  __bf16* wob  = (__bf16*)alloc((size_t)EMBED * EMBED * 2);
  __bf16* qb   = (__bf16*)alloc((size_t)MTOT * EMBED * 2);
  __bf16* kbm  = (__bf16*)alloc((size_t)MTOT * HDIM * 2);
  __bf16* vbm  = (__bf16*)alloc((size_t)MTOT * HDIM * 2);
  __bf16* vtb  = (__bf16*)alloc((size_t)MTOT * HDIM * 2);
  __bf16* ctxb = (__bf16*)alloc((size_t)MTOT * EMBED * 2);

  auto cvt = [&](const float* src, __bf16* dst, int n) {
    int thr = n / 8;
    f32_to_bf16_kernel<<<dim3((thr + 255) / 256), dim3(256), 0, stream>>>(src, dst, n);
  };
  cvt(x,  xb,  MTOT * EMBED);
  cvt(Wq, wqb, EMBED * EMBED);
  cvt(Wk, wkb, HDIM * EMBED);
  cvt(Wv, wvb, HDIM * EMBED);
  cvt(Wo, wob, EMBED * EMBED);

  // q = (x @ Wq^T + bq) * scaling  -> bf16 [MTOT, E]
  gemm_wmma_bf16<<<dim3(EMBED / 64, MTOT / 64), dim3(32), 0, stream>>>(
      xb, wqb, bq, QSCALE, qb, nullptr, MTOT, EMBED, EMBED);
  // k = x @ Wk^T -> bf16 [MTOT, 64]
  gemm_wmma_bf16<<<dim3(HDIM / 64, MTOT / 64), dim3(32), 0, stream>>>(
      xb, wkb, nullptr, 1.0f, kbm, nullptr, MTOT, HDIM, EMBED);
  // v = x @ Wv^T + bv -> bf16 [MTOT, 64]
  gemm_wmma_bf16<<<dim3(HDIM / 64, MTOT / 64), dim3(32), 0, stream>>>(
      xb, wvb, bv, 1.0f, vbm, nullptr, MTOT, HDIM, EMBED);

  transpose_v_kernel<<<dim3((MTOT * HDIM + 255) / 256), dim3(256), 0, stream>>>(
      vbm, vtb);

  mqa_attention_kernel<<<dim3(SEQ / 32, NHEADS, BSZ), dim3(32), 0, stream>>>(
      qb, kbm, vtb, ctxb);

  // out = ctx @ Wo^T + bo -> f32
  gemm_wmma_bf16<<<dim3(EMBED / 64, MTOT / 64), dim3(32), 0, stream>>>(
      ctxb, wob, bo, 1.0f, nullptr, out, MTOT, EMBED, EMBED);
}
